// GAT_79998060855611
// MI455X (gfx1250) — matrix operand order repaired and checked by reading, compiled, not run on hardware
//
#include <hip/hip_runtime.h>

typedef __attribute__((ext_vector_type(16))) _Float16 v16h;
typedef __attribute__((ext_vector_type(8)))  float    v8f;

#define NNODES 4096
#define NTILES 128           // NNODES / 32 column tiles

// ---------------------------------------------------------------------------
// Projection GEMM  C[N,O] = A[N,K] @ W[K,O] + bias   via v_wmma_f32_16x16x32_f16
// One wave computes one 16x16 output tile; K-loop in steps of 32.
// ---------------------------------------------------------------------------
__global__ __launch_bounds__(32)
void proj_wmma(const float* __restrict__ A, const float* __restrict__ W,
               const float* __restrict__ bias, float* __restrict__ C,
               int K, int O)
{
    const int lane = threadIdx.x;
    const int hi   = lane >> 4;
    const int l16  = lane & 15;
    const int row0 = blockIdx.x * 16;
    const int col0 = blockIdx.y * 16;
    const int k0   = hi * 8;

    v8f acc = {};
    const float* arow = A + (long)(row0 + l16) * K;
    for (int k = 0; k < K; k += 32) {
        v16h a;
        #pragma unroll
        for (int c = 0; c < 8; ++c) {
            a[c]     = (_Float16)arow[k + k0 + c];
            a[c + 8] = (_Float16)arow[k + 16 + k0 + c];
        }
        v16h b;
        #pragma unroll
        for (int c = 0; c < 16; ++c) {
            b[c] = (_Float16)W[(long)(k + ((c < 8) ? (k0 + c) : (16 + k0 + (c - 8)))) * O + col0 + l16];
        }
        acc = __builtin_amdgcn_wmma_f32_16x16x32_f16(false, a, false, b,
                                                     (short)0, acc, false, false);
    }
    const float bv = bias[col0 + l16];
    #pragma unroll
    for (int j = 0; j < 8; ++j) {
        int r = row0 + j + hi * 8;
        C[(long)r * O + col0 + l16] = acc[j] + bv;
    }
}

// ---------------------------------------------------------------------------
// Per-node attention logits: s_src[h,n], s_tgt[h,n]
// ---------------------------------------------------------------------------
__global__ void gat_scores(const float* __restrict__ proj,
                           const float* __restrict__ a_src,
                           const float* __restrict__ a_tgt,
                           float* __restrict__ s_src, float* __restrict__ s_tgt,
                           int H, int F)
{
    int idx = blockIdx.x * blockDim.x + threadIdx.x;   // n*H + h
    if (idx >= NNODES * H) return;
    int n = idx / H, h = idx % H;
    const float* p  = proj + ((long)n * H + h) * F;
    const float* as = a_src + h * F;
    const float* at = a_tgt + h * F;
    float ss = 0.f, st = 0.f;
    for (int f = 0; f < F; ++f) {
        float v = p[f];
        ss += v * as[f];
        st += v * at[f];
    }
    s_src[h * NNODES + n] = ss;
    s_tgt[h * NNODES + n] = st;
}

// ---------------------------------------------------------------------------
// Per-head bound[h] = max_n s_tgt[h,n]  (safe softmax shift, shift-invariant)
// ---------------------------------------------------------------------------
__global__ __launch_bounds__(256)
void head_max(const float* __restrict__ s_tgt, float* __restrict__ bound)
{
    __shared__ float red[256];
    const int h = blockIdx.x;
    float m = -3.4e38f;
    for (int n = threadIdx.x; n < NNODES; n += 256)
        m = fmaxf(m, s_tgt[h * NNODES + n]);
    red[threadIdx.x] = m;
    __syncthreads();
    for (int s = 128; s > 0; s >>= 1) {
        if (threadIdx.x < s)
            red[threadIdx.x] = fmaxf(red[threadIdx.x], red[threadIdx.x + s]);
        __syncthreads();
    }
    if (threadIdx.x == 0) bound[h] = red[0];
}

// ---------------------------------------------------------------------------
// Compress the additive mask (0 / -1e9) to 1 bit per edge:
//   bits[n][t] bit k  <=>  mask[n][t*32+k] == 0
// ---------------------------------------------------------------------------
__global__ void pack_mask(const float* __restrict__ mask, unsigned* __restrict__ bits)
{
    int idx = blockIdx.x * blockDim.x + threadIdx.x;   // n*NTILES + t
    if (idx >= NNODES * NTILES) return;
    const float* src = mask + (long)idx * 32;
    unsigned b = 0;
    #pragma unroll
    for (int k = 0; k < 32; ++k)
        b |= (src[k] == 0.0f) ? (1u << k) : 0u;
    bits[idx] = b;
}

// ---------------------------------------------------------------------------
// Pack V = proj[:,h,:] into WMMA-B f16 operand order:
//   packed[h][t][ft][lane][c] = proj[t*32 + (c<8 ? 8*(lane>=16)+c : 16+8*(lane>=16)+c-8), h*64 + ft*16 + lane%16]
// ---------------------------------------------------------------------------
__global__ void pack_v(const float* __restrict__ proj, _Float16* __restrict__ packed,
                       int H)
{
    int idx = blockIdx.x * blockDim.x + threadIdx.x;     // chunk id
    if (idx >= H * NTILES * 4 * 32) return;
    int lane = idx & 31;
    int ft   = (idx >> 5) & 3;
    int t    = (idx >> 7) & (NTILES - 1);
    int h    = idx >> 14;                                // 128*4*32 = 2^14
    int hi = lane >> 4, l16 = lane & 15;
    int HF = H * 64;
    const float* src = proj + h * 64 + ft * 16 + l16;
    v16h v;
    #pragma unroll
    for (int c = 0; c < 16; ++c)
        v[c] = (_Float16)src[(long)(t * 32 + ((c < 8) ? (hi * 8 + c) : (16 + hi * 8 + (c - 8)))) * HF];
    *(v16h*)(packed + (long)idx * 16) = v;
}

// ---------------------------------------------------------------------------
// Fused masked-softmax attention. Workgroup = 4 waves, one head, 64 rows.
// V tiles (32x64 f16, pre-packed) double-buffered in LDS via
// global_load_async_to_lds_b128 (ASYNCcnt), shared by all 4 waves, one
// barrier per tile. Mask consumed as a bitmask; softmax uses the precomputed
// per-row upper bound (exact by shift invariance).
// ---------------------------------------------------------------------------
__global__ __launch_bounds__(128)
void gat_attn(const _Float16* __restrict__ packedV,  // (H, 128, 4, 32, 16) f16
              const float* __restrict__ s_src,       // (H, N)
              const float* __restrict__ s_tgt,       // (H, N)
              const float* __restrict__ bound,       // (H)
              const unsigned* __restrict__ mbits,    // (N, NTILES)
              float* __restrict__ out,               // (N, H*64)
              int H)
{
    __shared__ _Float16 vbuf[2 * 2048];   // 2 x 4KB tiles
    const int tid  = threadIdx.x;
    const int wv   = tid >> 5;
    const int lane = tid & 31;
    const int hi   = lane >> 4;
    const int l16  = lane & 15;
    const int h    = blockIdx.y;
    const int row0 = blockIdx.x * 64 + wv * 16;
    const int myrow = row0 + l16;
    const int HF   = H * 64;
    const int k0   = hi * 8;

    const float* stg  = s_tgt + h * NNODES;
    const unsigned* brow = mbits + (long)myrow * NTILES;
    const float  ssrc = s_src[h * NNODES + myrow];
    float eb = ssrc + bound[h];
    const float nC = -fmaxf(eb, 0.2f * eb) * 1.44269504f;   // -log2(e)*rowbound

    const unsigned ldsbase = (unsigned)(unsigned long long)vbuf + (unsigned)tid * 32u;
    const _Float16* gbase  = packedV + (long)h * NTILES * 2048 + tid * 16;

    auto issue = [&](int t, int b) {
        unsigned long long ga = (unsigned long long)(gbase + (long)t * 2048);
        unsigned lds = ldsbase + (unsigned)b * 4096u;
        asm volatile("global_load_async_to_lds_b128 %0, %1, off"
                     :: "v"(lds), "v"(ga) : "memory");
        asm volatile("global_load_async_to_lds_b128 %0, %1, off offset:16"
                     :: "v"(lds), "v"(ga) : "memory");
    };

    float l_run = 0.f;
    v8f acc0 = {}, acc1 = {}, acc2 = {}, acc3 = {};

    issue(0, 0);
    for (int t = 0; t < NTILES; ++t) {
        // own part of tile t complete; barrier publishes it to all waves AND
        // fences everyone's step-(t-1) reads of the buffer issue(t+1) refills
        asm volatile("s_wait_asynccnt 0x0" ::: "memory");
        __syncthreads();
        if (t < NTILES - 1) issue(t + 1, (t + 1) & 1);

        // ---- scores for the 16 columns this lane owns (A-operand K mapping)
        const int cb = t * 32 + k0;
        float tv[16];
        *(float4*)&tv[0]  = *(const float4*)(stg + cb);
        *(float4*)&tv[4]  = *(const float4*)(stg + cb + 4);
        *(float4*)&tv[8]  = *(const float4*)(stg + cb + 16);
        *(float4*)&tv[12] = *(const float4*)(stg + cb + 20);
        const unsigned bits = brow[t];

        v16h a;
        float psum = 0.f;
        #pragma unroll
        for (int c = 0; c < 16; ++c) {
            float e = ssrc + tv[c];
            e = fmaxf(e, 0.2f * e);                 // leaky relu (slope 0.2)
            float p = exp2f(fmaf(e, 1.44269504f, nC));
            const int bit = (c < 8) ? (k0 + c) : (16 + k0 + (c - 8));
            p = (bits & (1u << bit)) ? p : 0.0f;    // additive -1e9 == hard zero
            psum += p;
            a[c] = (_Float16)p;
        }
        l_run += psum;

        // ---- B operands straight from LDS (pre-packed layout)
        const _Float16* vb = vbuf + (t & 1) * 2048;
        v16h b0 = *(const v16h*)(vb + (0 * 32 + lane) * 16);
        v16h b1 = *(const v16h*)(vb + (1 * 32 + lane) * 16);
        v16h b2 = *(const v16h*)(vb + (2 * 32 + lane) * 16);
        v16h b3 = *(const v16h*)(vb + (3 * 32 + lane) * 16);

        acc0 = __builtin_amdgcn_wmma_f32_16x16x32_f16(false, a, false, b0, (short)0, acc0, false, false);
        acc1 = __builtin_amdgcn_wmma_f32_16x16x32_f16(false, a, false, b1, (short)0, acc1, false, false);
        acc2 = __builtin_amdgcn_wmma_f32_16x16x32_f16(false, a, false, b2, (short)0, acc2, false, false);
        acc3 = __builtin_amdgcn_wmma_f32_16x16x32_f16(false, a, false, b3, (short)0, acc3, false, false);
    }

    // ---- finalize: divide each row by its softmax denominator, store
    float l_tot = l_run + __shfl_xor(l_run, 16);
    float lrcp  = 1.0f / l_tot;
    #pragma unroll
    for (int j = 0; j < 8; ++j) {
        float rj = __shfl(lrcp, j + hi * 8);       // C comp j -> row j + 8*hi
        int r = row0 + j + hi * 8;
        float* po = out + (long)r * HF + h * 64 + l16;
        po[0]  = acc0[j] * rj;
        po[16] = acc1[j] * rj;
        po[32] = acc2[j] * rj;
        po[48] = acc3[j] * rj;
    }
}

// ---------------------------------------------------------------------------
extern "C" void kernel_launch(void* const* d_in, const int* in_sizes, int n_in,
                              void* d_out, int out_size, void* d_ws, size_t ws_size,
                              hipStream_t stream)
{
    const float* x      = (const float*)d_in[0];
    const float* mask   = (const float*)d_in[1];
    const float* W0     = (const float*)d_in[2];
    const float* b0     = (const float*)d_in[3];
    const float* a_src0 = (const float*)d_in[4];
    const float* a_tgt0 = (const float*)d_in[5];
    const float* W1     = (const float*)d_in[6];
    const float* b1     = (const float*)d_in[7];
    const float* a_src1 = (const float*)d_in[8];
    const float* a_tgt1 = (const float*)d_in[9];

    float* ws    = (float*)d_ws;
    float* proj0 = ws;                       // 4096*512
    float* h0    = proj0 + NNODES * 512;     // 4096*512
    float* proj1 = h0    + NNODES * 512;     // 4096*64
    float* ssrc0 = proj1 + NNODES * 64;      // 8*4096
    float* stgt0 = ssrc0 + 8 * NNODES;
    float* ssrc1 = stgt0 + 8 * NNODES;       // 4096
    float* stgt1 = ssrc1 + NNODES;
    float* bound0 = stgt1 + NNODES;          // 8
    float* bound1 = bound0 + 8;              // 8 (padded)
    _Float16* packed0 = (_Float16*)(bound1 + 8);          // 8*128*128*16 halves
    _Float16* packed1 = packed0 + 8 * NTILES * 128 * 16;  // 1*128*128*16 halves
    unsigned* mbits   = (unsigned*)(packed1 + NTILES * 128 * 16);  // 4096*128

    // ---- shared precompute
    pack_mask<<<(NNODES * NTILES) / 256, 256, 0, stream>>>(mask, mbits);

    // ---- Layer 0 (H=8, F=64)
    proj_wmma<<<dim3(NNODES / 16, 512 / 16), 32, 0, stream>>>(x, W0, b0, proj0, 128, 512);
    gat_scores<<<(NNODES * 8 + 255) / 256, 256, 0, stream>>>(proj0, a_src0, a_tgt0,
                                                             ssrc0, stgt0, 8, 64);
    head_max<<<8, 256, 0, stream>>>(stgt0, bound0);
    pack_v<<<(8 * NTILES * 4 * 32) / 256, 256, 0, stream>>>(proj0, packed0, 8);
    gat_attn<<<dim3(NNODES / 64, 8), 128, 0, stream>>>(packed0, ssrc0, stgt0, bound0,
                                                       mbits, h0, 8);

    // ---- Layer 1 (H=1, F=64)
    proj_wmma<<<dim3(NNODES / 16, 64 / 16), 32, 0, stream>>>(h0, W1, b1, proj1, 512, 64);
    gat_scores<<<(NNODES + 255) / 256, 256, 0, stream>>>(proj1, a_src1, a_tgt1,
                                                         ssrc1, stgt1, 1, 64);
    head_max<<<1, 256, 0, stream>>>(stgt1, bound1);
    pack_v<<<(1 * NTILES * 4 * 32) / 256, 256, 0, stream>>>(proj1, packed1, 1);
    gat_attn<<<dim3(NNODES / 64, 1), 128, 0, stream>>>(packed1, ssrc1, stgt1, bound1,
                                                       mbits, (float*)d_out, 1);
}
